// GCNModel_29411936043506
// MI455X (gfx1250) — matrix operand order repaired and checked by reading, not compiled, not run on hardware
//
#include <hip/hip_runtime.h>
#include <hip/hip_bf16.h>

// ---------------------------------------------------------------------------
// GCN (3x GCNConv + BN + ReLU) -> mean pool -> MLP head, for MI455X (gfx1250).
// GEMMs via v_wmma_f32_16x16x32_bf16; scatter via global_atomic_add_f32.
// GEMM epilogue fused with the GCN self-loop init (agg = h*dinv^2 + bias).
// ---------------------------------------------------------------------------

typedef __attribute__((ext_vector_type(16))) __bf16 v16bf;
typedef __attribute__((ext_vector_type(8)))  float  v8f;

#define HDIM 64
#define NGRAPH 64
#define BN_EPS 1e-5f

// ------------------------------ utility --------------------------------------
__global__ void zero_f32(float* __restrict__ p, int n) {
    int i = blockIdx.x * blockDim.x + threadIdx.x;
    if (i < n) p[i] = 0.0f;
}

// deg[dst] += 1 per edge
__global__ void degree_kernel(const int* __restrict__ dst, float* __restrict__ deg, int nedges) {
    int e = blockIdx.x * blockDim.x + threadIdx.x;
    if (e < nedges) unsafeAtomicAdd(&deg[dst[e]], 1.0f);
}

// dinv = rsqrt(deg + 1)   (in place)
__global__ void rsqrt_kernel(float* __restrict__ deg, int n) {
    int i = blockIdx.x * blockDim.x + threadIdx.x;
    if (i < n) deg[i] = rsqrtf(deg[i] + 1.0f);
}

// --------------- GEMM + self-loop epilogue: H = X @ W^T ; A = H*dinv^2 + b ---
// X: [nrows, Cin] f32 row-major, W: [64, Cin] f32 row-major.
// Block = 256 threads = 8 waves; block tile = 32 rows x 64 cols;
// each wave computes one 16x16 tile with K-steps of 32 via WMMA bf16.
// Out-of-range rows are handled by CLAMPING the A-row pointer (each A row only
// feeds its own D row; stores are guarded only on the genuine tail tile), so
// both the k-loop and the common-path epilogue are branch/predication free.
__global__ void __launch_bounds__(256)
gemm_selfloop_wmma(const float* __restrict__ X, const float* __restrict__ W,
                   const float* __restrict__ dinv, const float* __restrict__ bias,
                   float* __restrict__ H, float* __restrict__ A,
                   int nrows, int Cin) {
    const int wave  = threadIdx.x >> 5;
    const int lane  = threadIdx.x & 31;
    const int rowTile = blockIdx.x * 2 + (wave >> 2);
    const int colTile = wave & 3;
    const int m0 = rowTile * 16;
    const int n0 = colTile * 16;
    if (m0 >= nrows) return;                 // uniform per wave

    const int mn    = lane & 15;             // A-row / B-col served by this lane
    const int khalf = lane >> 4;             // which K-half of the 32-wide step

    int row = m0 + mn;
    if (row >= nrows) row = nrows - 1;       // clamp; garbage rows never stored
    const float* arow = X + (size_t)row * Cin;
    const float* brow = W + (size_t)(n0 + mn) * Cin;   // B(k,n) = W[n][k]

    v8f acc = {};
    for (int k0 = 0; k0 < Cin; k0 += 32) {
        v16bf af, bf;
#pragma unroll
        for (int v = 0; v < 8; ++v) {
            // A layout: VGPR v holds K = 16*(v>=4) + 8*khalf + 2*(v&3) + {0,1}
            const int kA = k0 + ((v >= 4) ? 16 : 0) + 8 * khalf + 2 * (v & 3);
            const float2 a = *(const float2*)(arow + kA);
            af[2 * v]     = (__bf16)a.x;
            af[2 * v + 1] = (__bf16)a.y;
            // B layout: VGPR v holds K = 16*(v>=4) + 8*khalf + 2*(v&3) + {0,1}
            const int kB = k0 + ((v >= 4) ? 16 : 0) + 8 * khalf + 2 * (v & 3);
            const float2 b = *(const float2*)(brow + kB);
            bf[2 * v]     = (__bf16)b.x;
            bf[2 * v + 1] = (__bf16)b.y;
        }
        acc = __builtin_amdgcn_wmma_f32_16x16x32_bf16(
            /*neg_a=*/false, af, /*neg_b=*/false, bf,
            /*c_mod=*/(short)0, acc, /*reuse_a=*/false, /*reuse_b=*/false);
    }

    // C/D layout: lane n = lane&15, VGPR r -> row r + 8*(lane>>4)
    const int n   = lane & 15;
    const int mh  = lane >> 4;
    const int r0  = m0 + 8 * mh;                    // first row this lane stores
    const float bi = bias[n0 + n];
    float*       hp = H + (size_t)r0 * HDIM + n0 + n;
    float*       ap = A + (size_t)r0 * HDIM + n0 + n;
    const float* dp = dinv + r0;

    if (m0 + 16 <= nrows) {                          // common path: no guards
#pragma unroll
        for (int r = 0; r < 8; ++r) {
            const float h  = acc[r];
            const float di = dp[r];
            hp[r * HDIM] = h;
            ap[r * HDIM] = h * di * di + bi;
        }
    } else {                                         // tail tile only
#pragma unroll
        for (int r = 0; r < 8; ++r) {
            if (r0 + r < nrows) {
                const float h  = acc[r];
                const float di = dp[r];
                hp[r * HDIM] = h;
                ap[r * HDIM] = h * di * di + bi;
            }
        }
    }
}

// ---------------- edge scatter: agg[dst] += h[src] * dinv[src]*dinv[dst] -----
#define EPW 4   // edges per wave
__global__ void __launch_bounds__(256)
edge_scatter(const float* __restrict__ H, const int* __restrict__ src,
             const int* __restrict__ dst, const float* __restrict__ dinv,
             float* __restrict__ agg, int nedges) {
    const int wid  = (blockIdx.x * blockDim.x + threadIdx.x) >> 5;
    const int lane = threadIdx.x & 31;
    const int e0   = wid * EPW;
#pragma unroll
    for (int i = 0; i < EPW; ++i) {
        const int e = e0 + i;
        if (e >= nedges) return;
        const int s = src[e];
        const int d = dst[e];
        if (e + 1 < nedges)   // cover gather latency of the next edge's row
            __builtin_prefetch(H + (size_t)src[e + 1] * HDIM, 0, 0);
        const float w = dinv[s] * dinv[d];
        const float2 v = *(const float2*)(H + (size_t)s * HDIM + 2 * lane);
        float* a = agg + (size_t)d * HDIM + 2 * lane;
        unsafeAtomicAdd(a,     v.x * w);
        unsafeAtomicAdd(a + 1, v.y * w);
    }
}

// --------------------------- BatchNorm statistics ----------------------------
// sums/sumsq: [64]; block = 256 threads = 4 row-groups x 64 features.
__global__ void __launch_bounds__(256)
bn_reduce(const float* __restrict__ X, float* __restrict__ sums,
          float* __restrict__ sumsq, int nrows) {
    __shared__ float ssum[256];
    __shared__ float ssq[256];
    const int f = threadIdx.x & 63;
    const int g = threadIdx.x >> 6;
    float s = 0.0f, q = 0.0f;
    for (int r = blockIdx.x * 4 + g; r < nrows; r += gridDim.x * 4) {
        const float x = X[(size_t)r * HDIM + f];
        s += x;
        q += x * x;
    }
    ssum[threadIdx.x] = s;
    ssq[threadIdx.x]  = q;
    __syncthreads();
    if (threadIdx.x < 64) {
        s = ssum[f] + ssum[64 + f] + ssum[128 + f] + ssum[192 + f];
        q = ssq[f]  + ssq[64 + f]  + ssq[128 + f]  + ssq[192 + f];
        unsafeAtomicAdd(&sums[f], s);
        unsafeAtomicAdd(&sumsq[f], q);
    }
}

// sums -> scale, sumsq -> shift  (in place; 64 threads)
__global__ void bn_finalize(float* __restrict__ sums, float* __restrict__ sumsq,
                            const float* __restrict__ gamma, const float* __restrict__ beta,
                            float inv_n) {
    const int f = threadIdx.x;
    if (f >= 64) return;
    const float m  = sums[f] * inv_n;
    const float v  = sumsq[f] * inv_n - m * m;        // biased variance
    const float sc = gamma[f] * rsqrtf(v + BN_EPS);
    sums[f]  = sc;
    sumsq[f] = beta[f] - m * sc;
}

// y = relu(x*scale + shift)   (in place)
__global__ void bn_apply_relu(float* __restrict__ X, const float* __restrict__ scale,
                              const float* __restrict__ shift, int nrows) {
    int t = blockIdx.x * blockDim.x + threadIdx.x;
    int node = t >> 5;
    int lane = t & 31;
    if (node >= nrows) return;
    float2* p = (float2*)(X + (size_t)node * HDIM + 2 * lane);
    float2 v = *p;
    v.x = fmaxf(v.x * scale[2 * lane]     + shift[2 * lane],     0.0f);
    v.y = fmaxf(v.y * scale[2 * lane + 1] + shift[2 * lane + 1], 0.0f);
    *p = v;
}

// ------------------------------- pooling -------------------------------------
__global__ void pool_scatter(const float* __restrict__ H, const int* __restrict__ batch,
                             float* __restrict__ pooled, float* __restrict__ counts, int nrows) {
    const int t = blockIdx.x * blockDim.x + threadIdx.x;
    const int node = t >> 5;
    const int lane = t & 31;
    if (node >= nrows) return;
    const int b = batch[node];
    const float2 v = *(const float2*)(H + (size_t)node * HDIM + 2 * lane);
    float* p = pooled + (size_t)b * HDIM + 2 * lane;
    unsafeAtomicAdd(p,     v.x);
    unsafeAtomicAdd(p + 1, v.y);
    if (lane == 0) unsafeAtomicAdd(&counts[b], 1.0f);
}

// --------------------------------- head --------------------------------------
// one wave per graph: lane j computes relu(pooled_mean . lin1_w[j] + lin1_b[j]),
// then shuffle-reduce the lin2 dot product.
__global__ void head_kernel(const float* __restrict__ pooled, const float* __restrict__ counts,
                            const float* __restrict__ l1w, const float* __restrict__ l1b,
                            const float* __restrict__ l2w, const float* __restrict__ l2b,
                            float* __restrict__ out) {
    const int g = blockIdx.x;
    const int j = threadIdx.x;               // 0..31
    const float invc = 1.0f / fmaxf(counts[g], 1.0f);
    float acc = l1b[j];
#pragma unroll
    for (int k = 0; k < HDIM; ++k)
        acc += (pooled[g * HDIM + k] * invc) * l1w[j * HDIM + k];
    acc = fmaxf(acc, 0.0f) * l2w[j];
#pragma unroll
    for (int off = 16; off > 0; off >>= 1)
        acc += __shfl_down(acc, off, 32);
    if (j == 0) out[g] = acc + l2b[0];
}

// ------------------------------- launcher ------------------------------------
static inline size_t align_up(size_t x, size_t a) { return (x + a - 1) & ~(a - 1); }

extern "C" void kernel_launch(void* const* d_in, const int* in_sizes, int n_in,
                              void* d_out, int out_size, void* d_ws, size_t ws_size,
                              hipStream_t stream) {
    const float* x      = (const float*)d_in[0];
    const int*   ei     = (const int*)d_in[1];
    const int*   batch  = (const int*)d_in[2];
    const float* W[3]   = { (const float*)d_in[3],  (const float*)d_in[7],  (const float*)d_in[11] };
    const float* B[3]   = { (const float*)d_in[4],  (const float*)d_in[8],  (const float*)d_in[12] };
    const float* Ga[3]  = { (const float*)d_in[5],  (const float*)d_in[9],  (const float*)d_in[13] };
    const float* Be[3]  = { (const float*)d_in[6],  (const float*)d_in[10], (const float*)d_in[14] };
    const float* l1w    = (const float*)d_in[15];
    const float* l1b    = (const float*)d_in[16];
    const float* l2w    = (const float*)d_in[17];
    const float* l2b    = (const float*)d_in[18];
    float*       out    = (float*)d_out;

    const int N    = in_sizes[2];            // nodes
    const int E    = in_sizes[1] / 2;        // edges
    const int Cin0 = in_sizes[0] / N;        // 128
    const int* src = ei;
    const int* dst = ei + E;

    // workspace carve-up
    char* base = (char*)d_ws;
    size_t off = 0;
    const size_t hbytes = align_up((size_t)N * HDIM * sizeof(float), 256);
    float* hA     = (float*)(base + off); off += hbytes;   // GEMM output h
    float* hB     = (float*)(base + off); off += hbytes;   // aggregation buffer
    float* dinv   = (float*)(base + off); off += align_up((size_t)N * sizeof(float), 256);
    float* sums   = (float*)(base + off); off += 256;
    float* sumsq  = (float*)(base + off); off += 256;
    float* pooled = (float*)(base + off); off += align_up((size_t)NGRAPH * HDIM * sizeof(float), 256);
    float* counts = (float*)(base + off); off += 256;
    (void)ws_size; (void)n_in; (void)out_size;

    // ---- degrees (shared by all three layers) ----
    zero_f32<<<(N + 255) / 256, 256, 0, stream>>>(dinv, N);
    degree_kernel<<<(E + 255) / 256, 256, 0, stream>>>(dst, dinv, E);
    rsqrt_kernel<<<(N + 255) / 256, 256, 0, stream>>>(dinv, N);

    const int gemm_blocks  = (N + 31) / 32;
    const int node_waves   = (N * 32 + 255) / 256;           // one wave per node
    const int edge_blocks  = (E + 8 * EPW - 1) / (8 * EPW);  // 8 waves/block, EPW edges/wave
    const int bn_blocks    = 120;

    const float* hin = x;
    int Cin = Cin0;
    for (int l = 0; l < 3; ++l) {
        // h = hin @ W^T   and   agg = h*dinv^2 + bias   (fused epilogue)
        gemm_selfloop_wmma<<<gemm_blocks, 256, 0, stream>>>(hin, W[l], dinv, B[l], hA, hB, N, Cin);
        edge_scatter<<<edge_blocks, 256, 0, stream>>>(hA, src, dst, dinv, hB, E);
        zero_f32<<<1, 128, 0, stream>>>(sums, 64);
        zero_f32<<<1, 128, 0, stream>>>(sumsq, 64);
        bn_reduce<<<bn_blocks, 256, 0, stream>>>(hB, sums, sumsq, N);
        bn_finalize<<<1, 64, 0, stream>>>(sums, sumsq, Ga[l], Be[l], 1.0f / (float)N);
        bn_apply_relu<<<node_waves, 256, 0, stream>>>(hB, sums, sumsq, N);
        hin = hB;
        Cin = HDIM;
    }

    // ---- pooling + head ----
    zero_f32<<<(NGRAPH * HDIM + 255) / 256, 256, 0, stream>>>(pooled, NGRAPH * HDIM);
    zero_f32<<<1, 128, 0, stream>>>(counts, NGRAPH);
    pool_scatter<<<node_waves, 256, 0, stream>>>(hB, batch, pooled, counts, N);
    head_kernel<<<NGRAPH, 32, 0, stream>>>(pooled, counts, l1w, l1b, l2w, l2b, out);
}
